// Decoder_52690658787452
// MI455X (gfx1250) — matrix operand order repaired and checked by reading, not compiled, not run on hardware
//
#include <hip/hip_runtime.h>
#include <hip/hip_bf16.h>
#include <math.h>

#define BB 64
#define TT 128
#define EE 256
#define HH 512
#define SS 1024

typedef __attribute__((ext_vector_type(16))) __bf16 v16bf;
typedef __attribute__((ext_vector_type(8)))  float  v8f;

union Frag { unsigned int u[8]; v16bf v; };

__device__ __forceinline__ unsigned short f2bf(float a) {
  unsigned int ua = __float_as_uint(a);
  return (unsigned short)((ua + 0x7fffu + ((ua >> 16) & 1u)) >> 16);  // RNE bf16
}

// Load 8 consecutive bf16 (16 B) into u[0..3]
__device__ __forceinline__ void load8bf(const unsigned short* p, unsigned int* u) {
  uint4 w = *(const uint4*)p;
  u[0] = w.x; u[1] = w.y; u[2] = w.z; u[3] = w.w;
}

// Load 16 consecutive bf16 (32 B) into u[0..7]
__device__ __forceinline__ void loadB16(const unsigned short* p, unsigned int* u) {
  const uint4* q = (const uint4*)p;
  uint4 w0 = q[0], w1 = q[1];
  u[0] = w0.x; u[1] = w0.y; u[2] = w0.z; u[3] = w0.w;
  u[4] = w1.x; u[5] = w1.y; u[6] = w1.z; u[7] = w1.w;
}

__device__ __forceinline__ float sigmoidf_(float x) { return 1.f / (1.f + __expf(-x)); }

// ---------------------------------------------------------------------------
// One-time per call: convert weights (transposed) and all of x to bf16.
//  W4t  [2048][768] = [W_ih ; W_hh]^T   (gates GEMM, K=768)
//  Woutt[512][1024] = W_out^T           (out GEMM,  K=1024)
//  Waxt [512][768]  = W_ax^T            (xa GEMM,   K=768)
//  xbf  [B,T,E]     = embedded_inputs   (bf16, same layout)
// ---------------------------------------------------------------------------
__global__ void k_convert(const float* __restrict__ Wih, const float* __restrict__ Whh,
                          const float* __restrict__ Wout, const float* __restrict__ Wax,
                          const float* __restrict__ emb,
                          unsigned short* __restrict__ W4t,
                          unsigned short* __restrict__ Woutt,
                          unsigned short* __restrict__ Waxt,
                          unsigned short* __restrict__ xbf) {
  int i = blockIdx.x * blockDim.x + threadIdx.x;
  const int N1 = 2048 * 768, N2 = 512 * 1024, N3 = 512 * 768, N4 = BB * TT * EE;
  if (i < N1) {
    int n = i / 768, k = i - n * 768;
    float v = (k < 256) ? Wih[k * 2048 + n] : Whh[(k - 256) * 2048 + n];
    W4t[i] = f2bf(v);
  } else if (i < N1 + N2) {
    int j = i - N1;
    int n = j / 1024, k = j - n * 1024;
    Woutt[j] = f2bf(Wout[k * 512 + n]);
  } else if (i < N1 + N2 + N3) {
    int j = i - N1 - N2;
    int n = j / 768, k = j - n * 768;
    Waxt[j] = f2bf(Wax[k * 512 + n]);
  } else if (i < N1 + N2 + N3 + N4) {
    int j = i - N1 - N2 - N3;
    xbf[j] = f2bf(emb[j]);
  }
}

__global__ void k_init(const float* __restrict__ h0, const float* __restrict__ c0,
                       float* __restrict__ hs, float* __restrict__ cs,
                       unsigned short* __restrict__ hbf0) {
  int i = blockIdx.x * blockDim.x + threadIdx.x;
  if (i < BB * HH) {
    hs[i]   = h0[i];
    cs[i]   = c0[i];
    hbf0[i] = f2bf(h0[i]);
  }
}

// ---------------------------------------------------------------------------
// LSTM cell: gates = [x_t | h_prev][64,768] @ W4t^T + biases, fused pointwise.
// Grid (32 htiles, 4 mtiles), 128 threads = 4 waves (one per gate i/f/g/o).
// All GEMM operands are pre-converted bf16; inner loop is pure loads + WMMA.
// h bf16 state is ping-ponged (all blocks read all of h_prev while writing h_t).
// ---------------------------------------------------------------------------
__global__ void k_lstm_cell(const unsigned short* __restrict__ xbf,
                            const unsigned short* __restrict__ hbf_cur,
                            float* __restrict__ cst, float* __restrict__ hnext,
                            unsigned short* __restrict__ hbf_nxt,
                            const unsigned short* __restrict__ W4t,
                            const float* __restrict__ bih, const float* __restrict__ bhh, int t) {
  __shared__ float lds[4][16][16];
  const int lane  = threadIdx.x & 31;
  const int gate  = threadIdx.x >> 5;
  const int htile = blockIdx.x;
  const int m0    = blockIdx.y * 16;
  const int n     = gate * 512 + htile * 16 + (lane & 15);   // B/C column
  const int arow  = m0 + (lane & 15);                        // A row (batch)
  const int kbA   = (lane < 16) ? 0 : 16;
  const int kbB   = (lane < 16) ? 0 : 16;
  const unsigned short* rowX = xbf + ((size_t)arow * TT + t) * EE;  // A cols [0,256)
  const unsigned short* rowH = hbf_cur + (size_t)arow * HH - 256;   // A cols [256,768)
  const unsigned short* wrow = W4t + (size_t)n * 768;

  v8f acc = {};
  #pragma unroll 4
  for (int kk = 0; kk < 768; kk += 32) {
    Frag a, b;
    int g0 = kk + kbA;
    int g1 = g0 + 8;
    load8bf((g0 < 256 ? rowX : rowH) + g0, a.u);
    load8bf((g1 < 256 ? rowX : rowH) + g1, a.u + 4);
    loadB16(wrow + kk + kbB, b.u);
    acc = __builtin_amdgcn_wmma_f32_16x16x32_bf16(false, a.v, false, b.v, (short)0, acc,
                                                  false, false);
  }
  const float bias = bih[n] + bhh[n];
  const int r0 = (lane >> 4) * 8;
  #pragma unroll
  for (int r = 0; r < 8; ++r) lds[gate][r0 + r][lane & 15] = acc[r] + bias;
  __syncthreads();

  for (int e = threadIdx.x; e < 256; e += 128) {
    int rr = e >> 4, cc = e & 15;
    int b_ = m0 + rr;
    int hc = htile * 16 + cc;
    float gi = lds[0][rr][cc];
    float gf = lds[1][rr][cc];
    float gg = lds[2][rr][cc];
    float go = lds[3][rr][cc];
    float cp = cst[b_ * HH + hc];
    float ct = sigmoidf_(gf) * cp + sigmoidf_(gi) * tanhf(gg);
    float ht = sigmoidf_(go) * tanhf(ct);
    cst[b_ * HH + hc]     = ct;
    hnext[b_ * HH + hc]   = ht;
    hbf_nxt[b_ * HH + hc] = f2bf(ht);
  }
}

// ---------------------------------------------------------------------------
// Dot attention for one batch row per block. Context is L2-resident (134 MB of
// 192 MB L2); the GEMVs are bandwidth-bound so fp32 VALU is the right path.
// ---------------------------------------------------------------------------
__global__ void k_attention(const float* __restrict__ ctx, const float* __restrict__ h,
                            float* __restrict__ attn_out, float* __restrict__ actx,
                            unsigned short* __restrict__ actxbf) {
  __shared__ float sh[HH];
  __shared__ float sp[SS];
  __shared__ float red[256];
  const int b = blockIdx.x;
  const int tid = threadIdx.x;
  for (int i = tid; i < HH; i += 256) sh[i] = h[b * HH + i];
  __syncthreads();

  const float NEGINF = -__builtin_inff();
  for (int s = tid; s < SS; s += 256) {
    const float* row = ctx + ((size_t)b * SS + s) * HH;
    if (s + 256 < SS) __builtin_prefetch(row + (size_t)256 * HH, 0, 1);  // global_prefetch_b8
    float dot = 0.f, rs = 0.f;
    #pragma unroll 4
    for (int k = 0; k < HH; k += 4) {
      float4 v = *(const float4*)(row + k);
      dot += v.x * sh[k] + v.y * sh[k + 1] + v.z * sh[k + 2] + v.w * sh[k + 3];
      rs  += v.x + v.y + v.z + v.w;
    }
    sp[s] = (rs == 0.f) ? NEGINF : dot;   // mask rows whose mean == 0
  }
  __syncthreads();

  float lm = NEGINF;
  for (int s = tid; s < SS; s += 256) lm = fmaxf(lm, sp[s]);
  red[tid] = lm; __syncthreads();
  for (int o = 128; o > 0; o >>= 1) { if (tid < o) red[tid] = fmaxf(red[tid], red[tid + o]); __syncthreads(); }
  const float mx = red[0];
  __syncthreads();

  float lsum = 0.f;
  for (int s = tid; s < SS; s += 256) { float e_ = __expf(sp[s] - mx); sp[s] = e_; lsum += e_; }
  red[tid] = lsum; __syncthreads();
  for (int o = 128; o > 0; o >>= 1) { if (tid < o) red[tid] += red[tid + o]; __syncthreads(); }
  const float inv = 1.f / red[0];
  __syncthreads();

  for (int s = tid; s < SS; s += 256) {
    float p = sp[s] * inv;
    sp[s] = p;
    attn_out[(size_t)b * SS + s] = p;
  }
  __syncthreads();

  // attn_ctx[h] = sum_s p[s] * ctx[b,s,h]; coalesced across h.
  float a1 = 0.f, a2 = 0.f;
  const float* base = ctx + (size_t)b * SS * HH;
  for (int s = 0; s < SS; ++s) {
    float p = sp[s];
    a1 += p * base[(size_t)s * HH + tid];
    a2 += p * base[(size_t)s * HH + tid + 256];
  }
  actx[b * HH + tid]         = a1;
  actx[b * HH + tid + 256]   = a2;
  actxbf[b * HH + tid]       = f2bf(a1);
  actxbf[b * HH + tid + 256] = f2bf(a2);
}

// ---------------------------------------------------------------------------
// out = [actx|h_t] @ W_out + b_out (z=0, K=1024)  -> d_out outputs[t]
// xa  = [actx|x_t] @ W_ax  + b_ax  (z=1, K=768)   -> ws
// Grid (8 ntile-groups, 4 mtiles, 2 gemms), 128 threads = 4 waves.
// ---------------------------------------------------------------------------
__global__ void k_outgemms(const unsigned short* __restrict__ actxbf,
                           const unsigned short* __restrict__ hbf_nxt,
                           const unsigned short* __restrict__ xbf,
                           const unsigned short* __restrict__ Woutt,
                           const unsigned short* __restrict__ Waxt,
                           const float* __restrict__ bout, const float* __restrict__ bax,
                           float* __restrict__ out_t, float* __restrict__ xa, int t) {
  const int lane  = threadIdx.x & 31;
  const int wave  = threadIdx.x >> 5;
  const int ntile = blockIdx.x * 4 + wave;
  const int n     = ntile * 16 + (lane & 15);
  const int m0    = blockIdx.y * 16;
  const int arow  = m0 + (lane & 15);
  const int z     = blockIdx.z;
  const int K     = (z == 0) ? 1024 : 768;
  const unsigned short* rowA0 = actxbf + (size_t)arow * HH;          // cols [0,512)
  const unsigned short* rowA1 = (z == 0)
      ? (hbf_nxt + (size_t)arow * HH - 512)
      : (xbf + ((size_t)arow * TT + t) * EE - 512);
  const unsigned short* wrow = ((z == 0) ? Woutt : Waxt) + (size_t)n * K;
  const int kbA = (lane < 16) ? 0 : 16;
  const int kbB = (lane < 16) ? 0 : 16;

  v8f acc = {};
  #pragma unroll 4
  for (int kk = 0; kk < K; kk += 32) {
    Frag a, b;
    int g0 = kk + kbA;
    int g1 = g0 + 8;
    load8bf((g0 < 512 ? rowA0 : rowA1) + g0, a.u);
    load8bf((g1 < 512 ? rowA0 : rowA1) + g1, a.u + 4);
    loadB16(wrow + kk + kbB, b.u);
    acc = __builtin_amdgcn_wmma_f32_16x16x32_bf16(false, a.v, false, b.v, (short)0, acc,
                                                  false, false);
  }
  const float bias = ((z == 0) ? bout : bax)[n];
  float* dst = (z == 0) ? out_t : xa;
  const int r0 = (lane >> 4) * 8;
  #pragma unroll
  for (int r = 0; r < 8; ++r) {
    int m = m0 + r0 + r;
    dst[(size_t)m * HH + n] = acc[r] + bias;
  }
}

// p_gen[b] = sigmoid([actx|h|c|xa] . W_pg + b_pg); tiny matvec -> block reduce.
__global__ void k_pgen(const float* __restrict__ actx, const float* __restrict__ hcur,
                       const float* __restrict__ cst, const float* __restrict__ xa,
                       const float* __restrict__ Wpg, const float* __restrict__ bpg,
                       float* __restrict__ pg_t) {
  __shared__ float red[256];
  const int b = blockIdx.x, tid = threadIdx.x;
  float s = 0.f;
  for (int k = tid; k < 2048; k += 256) {
    float a;
    if (k < 512)       a = actx[b * HH + k];
    else if (k < 1024) a = hcur[b * HH + (k - 512)];
    else if (k < 1536) a = cst[b * HH + (k - 1024)];
    else               a = xa[b * HH + (k - 1536)];
    s += a * Wpg[k];
  }
  red[tid] = s; __syncthreads();
  for (int o = 128; o > 0; o >>= 1) { if (tid < o) red[tid] += red[tid + o]; __syncthreads(); }
  if (tid == 0) pg_t[b] = sigmoidf_(red[0] + bpg[0]);
}

__global__ void k_final(const float* __restrict__ hs, const float* __restrict__ cs,
                        float* __restrict__ outh, float* __restrict__ outc) {
  int i = blockIdx.x * blockDim.x + threadIdx.x;
  if (i < BB * HH) { outh[i] = hs[i]; outc[i] = cs[i]; }
}

extern "C" void kernel_launch(void* const* d_in, const int* in_sizes, int n_in,
                              void* d_out, int out_size, void* d_ws, size_t ws_size,
                              hipStream_t stream) {
  (void)in_sizes; (void)n_in; (void)out_size; (void)ws_size;
  const float* emb  = (const float*)d_in[0];
  const float* h0   = (const float*)d_in[1];
  const float* c0   = (const float*)d_in[2];
  const float* ctx  = (const float*)d_in[3];
  const float* Wih  = (const float*)d_in[4];
  const float* bih  = (const float*)d_in[5];
  const float* Whh  = (const float*)d_in[6];
  const float* bhh  = (const float*)d_in[7];
  const float* Wout = (const float*)d_in[8];
  const float* bout = (const float*)d_in[9];
  const float* Wax  = (const float*)d_in[10];
  const float* bax  = (const float*)d_in[11];
  const float* Wpg  = (const float*)d_in[12];
  const float* bpg  = (const float*)d_in[13];

  char* ws = (char*)d_ws;
  size_t off = 0;
  auto alloc = [&](size_t bytes) {
    void* p = ws + off;
    off = (off + bytes + 255) & ~(size_t)255;
    return p;
  };
  unsigned short* W4t    = (unsigned short*)alloc(2048ull * 768 * 2);
  unsigned short* Woutt  = (unsigned short*)alloc(512ull * 1024 * 2);
  unsigned short* Waxt   = (unsigned short*)alloc(512ull * 768 * 2);
  unsigned short* xbf    = (unsigned short*)alloc((size_t)BB * TT * EE * 2);
  unsigned short* hbf0   = (unsigned short*)alloc((size_t)BB * HH * 2);
  unsigned short* hbf1   = (unsigned short*)alloc((size_t)BB * HH * 2);
  unsigned short* actxbf = (unsigned short*)alloc((size_t)BB * HH * 2);
  float* hsf  = (float*)alloc((size_t)BB * HH * 4);
  float* cst  = (float*)alloc((size_t)BB * HH * 4);
  float* actx = (float*)alloc((size_t)BB * HH * 4);
  float* xa   = (float*)alloc((size_t)BB * HH * 4);
  unsigned short* hbf[2] = {hbf0, hbf1};

  float* o = (float*)d_out;
  float* out_outputs = o;                                  // [T,B,H]
  float* out_h    = o + (size_t)TT * BB * HH;              // [B,H]
  float* out_c    = out_h + (size_t)BB * HH;               // [B,H]
  float* out_attn = out_c + (size_t)BB * HH;               // [T,B,S]
  float* out_pg   = out_attn + (size_t)TT * BB * SS;       // [T,B,1]

  {
    const int total = 2048 * 768 + 512 * 1024 + 512 * 768 + BB * TT * EE;
    k_convert<<<(total + 255) / 256, 256, 0, stream>>>(Wih, Whh, Wout, Wax, emb,
                                                       W4t, Woutt, Waxt, xbf);
    k_init<<<(BB * HH + 255) / 256, 256, 0, stream>>>(h0, c0, hsf, cst, hbf[0]);
  }

  for (int t = 0; t < TT; ++t) {
    const int cur = t & 1, nxt = cur ^ 1;
    k_lstm_cell<<<dim3(32, 4), 128, 0, stream>>>(xbf, hbf[cur], cst, hsf, hbf[nxt],
                                                 W4t, bih, bhh, t);
    k_attention<<<BB, 256, 0, stream>>>(ctx, hsf, out_attn + (size_t)t * BB * SS,
                                        actx, actxbf);
    k_outgemms<<<dim3(8, 4, 2), 128, 0, stream>>>(actxbf, hbf[nxt], xbf, Woutt, Waxt,
                                                  bout, bax,
                                                  out_outputs + (size_t)t * BB * HH, xa, t);
    k_pgen<<<BB, 256, 0, stream>>>(actx, hsf, cst, xa, Wpg, bpg, out_pg + (size_t)t * BB);
  }
  k_final<<<(BB * HH + 255) / 256, 256, 0, stream>>>(hsf, cst, out_h, out_c);
}
